// MultiHeadAttention_13245679140887
// MI455X (gfx1250) — matrix operand order repaired and checked by reading, not compiled, not run on hardware
//
#include <hip/hip_runtime.h>
#include <hip/hip_bf16.h>

// ---------------------------------------------------------------------------
// MI455X (gfx1250) multi-head attention, bf16 WMMA (f32 accumulate)
// ---------------------------------------------------------------------------

#define Bsz 2
#define Ssz 2048
#define Esz 1024
#define Hsz 16
#define DKsz 64

// Flip to 0 if the async-to-LDS inline asm fails to assemble.
#ifndef ATTN_ASYNC
#define ATTN_ASYNC 1
#endif

typedef __attribute__((ext_vector_type(16))) __bf16 v16bf;
typedef __attribute__((ext_vector_type(8)))  float  v8f;

__device__ __forceinline__ v8f wmma_bf16(v16bf a, v16bf b, v8f c) {
  return __builtin_amdgcn_wmma_f32_16x16x32_bf16(
      false, a, false, b, (short)0, c, false, false);
}

// A-fragment K index for VGPR-pair v (0..7), half = lane/16:
// K = half*16 + v*2   (pairs K, K+1 share one 32-bit reg)
__device__ __forceinline__ int a_kbase(int v, int half) {
  return (half << 4) + (v << 1);
}

__device__ __forceinline__ unsigned pack_bf16x2(float a, float b) {
  union { __bf16 h[2]; unsigned u; } un;
  un.h[0] = (__bf16)a; un.h[1] = (__bf16)b;
  return un.u;
}

// ---------------------------------------------------------------------------
// Projection: Out[b,h,s,dk] = bf16( scale * (X[m,:] . W[n,:] + bias[n]) )
// Block tile 128(M) x 64(N), K-step 64.
// ---------------------------------------------------------------------------
#define PBM 128
#define PBN 64
#define PBK 64

__global__ __launch_bounds__(256) void proj_qkv_kernel(
    const float* __restrict__ X, const float* __restrict__ W,
    const float* __restrict__ bias, __bf16* __restrict__ Out, float scale) {
  __shared__ __bf16 sX[PBM][PBK];  // 16 KB
  __shared__ __bf16 sW[PBN][PBK];  // 8 KB

  const int tid  = threadIdx.x;
  const int wave = tid >> 5;
  const int lane = tid & 31;
  const int half = lane >> 4;
  const int l16  = lane & 15;
  const int m0   = blockIdx.x * PBM;
  const int n0   = blockIdx.y * PBN;

  const float4* X4 = (const float4*)X;   // Esz/4 float4 per row
  const float4* W4 = (const float4*)W;

  v8f acc[4] = {};

  for (int k0 = 0; k0 < Esz; k0 += PBK) {
    // ---- batched global loads (clauseable), then convert+store ----
    float4 xb[8];
#pragma unroll
    for (int i = 0; i < 8; ++i) {            // 128 rows x 16 float4 / 256 thr
      int idx = tid + i * 256;
      int r = idx >> 4, c4 = idx & 15;
      xb[i] = X4[(size_t)(m0 + r) * (Esz / 4) + (k0 >> 2) + c4];
    }
    float4 wb[4];
#pragma unroll
    for (int i = 0; i < 4; ++i) {            // 64 rows x 16 float4 / 256 thr
      int idx = tid + i * 256;
      int r = idx >> 4, c4 = idx & 15;
      wb[i] = W4[(size_t)(n0 + r) * (Esz / 4) + (k0 >> 2) + c4];
    }
#pragma unroll
    for (int i = 0; i < 8; ++i) {
      int idx = tid + i * 256;
      int r = idx >> 4, c4 = idx & 15;
      uint2 pk;
      pk.x = pack_bf16x2(xb[i].x, xb[i].y);
      pk.y = pack_bf16x2(xb[i].z, xb[i].w);
      *(uint2*)&sX[r][c4 * 4] = pk;          // ds_store_b64
    }
#pragma unroll
    for (int i = 0; i < 4; ++i) {
      int idx = tid + i * 256;
      int r = idx >> 4, c4 = idx & 15;
      uint2 pk;
      pk.x = pack_bf16x2(wb[i].x, wb[i].y);
      pk.y = pack_bf16x2(wb[i].z, wb[i].w);
      *(uint2*)&sW[r][c4 * 4] = pk;
    }
    __syncthreads();

    const unsigned* arow = (const unsigned*)&sX[wave * 16 + l16][0];
#pragma unroll
    for (int f = 0; f < 2; ++f) {            // K = f*32 .. f*32+31
      v16bf a;
#pragma unroll
      for (int v = 0; v < 8; ++v)
        ((unsigned*)&a)[v] = arow[(f * 32 + a_kbase(v, half)) >> 1];
#pragma unroll
      for (int sub = 0; sub < 4; ++sub) {
        v16bf bf;
        const unsigned* rowp = (const unsigned*)&sW[sub * 16 + l16][0];
#pragma unroll
        for (int v = 0; v < 8; ++v)
          ((unsigned*)&bf)[v] = rowp[f * 16 + (half << 3) + v];
        acc[sub] = wmma_bf16(a, bf, acc[sub]);
      }
    }
    __syncthreads();
  }

  // epilogue: bias, scale, head-split store [B,H,S,DK] bf16
#pragma unroll
  for (int sub = 0; sub < 4; ++sub) {
    int n  = n0 + sub * 16 + l16;
    int h  = n >> 6;
    int dk = n & 63;
    float bv = bias[n];
#pragma unroll
    for (int r = 0; r < 8; ++r) {
      int m = m0 + wave * 16 + r + half * 8;
      int b = m >> 11;
      int s = m & (Ssz - 1);
      float val = (acc[sub][r] + bv) * scale;
      Out[(((size_t)(b * Hsz + h)) * Ssz + s) * DKsz + dk] = (__bf16)val;
    }
  }
}

// ---------------------------------------------------------------------------
// Flash attention, key blocks of 64, faithful mask (value 1e-9, not -inf).
// Q pre-scaled by 1/sqrt(DK). One wave owns 16 query rows.
// ---------------------------------------------------------------------------
#define KBLK 64

__global__ __launch_bounds__(256) void attn_kernel(
    const __bf16* __restrict__ Q, const __bf16* __restrict__ K,
    const __bf16* __restrict__ V, const int* __restrict__ mask,
    __bf16* __restrict__ O) {
  __shared__ __bf16 sK[KBLK][DKsz];       // 8 KB  row-major [key][dim]
  __shared__ __bf16 sVt[DKsz][KBLK];      // 8 KB  transposed [dim][key]
  __shared__ __bf16 sP[8][16][KBLK];      // 16 KB per-wave P staging
  __shared__ int smask[KBLK];

  const int tid  = threadIdx.x;
  const int wave = tid >> 5;
  const int lane = tid & 31;
  const int half = lane >> 4;
  const int l16  = lane & 15;
  const int bh   = blockIdx.y;
  const int b    = bh >> 4;
  const int h    = bh & 15;
  const int q0   = blockIdx.x * 128;
  const size_t head_base = (size_t)bh * Ssz * DKsz;

#if ATTN_ASYNC
  const unsigned sK_lds =
      (unsigned)(unsigned long long)(__attribute__((address_space(3))) __bf16*)&sK[0][0];
#endif

  // Q A-fragments (16 rows x 64 dims = two 16x32 frags) in registers
  v16bf aq[2];
  {
    const unsigned* qrow =
        (const unsigned*)(Q + head_base + (size_t)(q0 + wave * 16 + l16) * DKsz);
#pragma unroll
    for (int f = 0; f < 2; ++f)
#pragma unroll
      for (int v = 0; v < 8; ++v)
        ((unsigned*)&aq[f])[v] = qrow[(f * 32 + a_kbase(v, half)) >> 1];
  }

  v8f o[4] = {};
  float mrow[8], lrow[8];
#pragma unroll
  for (int r = 0; r < 8; ++r) { mrow[r] = -3.0e38f; lrow[r] = 0.0f; }

  for (int k0 = 0; k0 < Ssz; k0 += KBLK) {
    // ---- K tile: async copy global -> LDS (ASYNCcnt path) ----
#pragma unroll
    for (int i = 0; i < 2; ++i) {            // 64x64 bf16 = 512 x 16B chunks
      int idx = tid + i * 256;
      int kr = idx >> 3, c8 = idx & 7;
      const __bf16* gp = K + head_base + (size_t)(k0 + kr) * DKsz + c8 * 8;
#if ATTN_ASYNC
      unsigned laddr = sK_lds + (unsigned)(kr * DKsz + c8 * 8) * 2u;
      asm volatile("global_load_async_to_lds_b128 %0, %1, off"
                   :: "v"(laddr), "v"((unsigned long long)gp) : "memory");
#else
      *(uint4*)&sK[kr][c8 * 8] = *(const uint4*)gp;
#endif
    }
    // ---- V tile: batched vector loads, transposed scatter into sVt ----
    uint4 vb[2];
#pragma unroll
    for (int i = 0; i < 2; ++i) {
      int idx = tid + i * 256;
      int kr = idx >> 3, c8 = idx & 7;
      vb[i] = *(const uint4*)(V + head_base + (size_t)(k0 + kr) * DKsz + c8 * 8);
    }
#pragma unroll
    for (int i = 0; i < 2; ++i) {
      int idx = tid + i * 256;
      int kr = idx >> 3, c8 = idx & 7;
      const unsigned short* us = (const unsigned short*)&vb[i];
#pragma unroll
      for (int j = 0; j < 8; ++j)
        *(unsigned short*)&sVt[c8 * 8 + j][kr] = us[j];
    }
    if (tid < KBLK) smask[tid] = mask[b * Ssz + k0 + tid];
#if ATTN_ASYNC
    asm volatile("s_wait_asynccnt 0x0" ::: "memory");
#endif
    __syncthreads();

    // ---- scores: four 16x16 tiles (64 keys), 2 WMMAs each over DK=64 ----
    v8f sc[4] = {};
#pragma unroll
    for (int sub = 0; sub < 4; ++sub) {
      const unsigned* rowp = (const unsigned*)&sK[sub * 16 + l16][0];
#pragma unroll
      for (int f = 0; f < 2; ++f) {
        v16bf bf;
#pragma unroll
        for (int v = 0; v < 8; ++v)
          ((unsigned*)&bf)[v] = rowp[f * 16 + (half << 3) + v];
        sc[sub] = wmma_bf16(aq[f], bf, sc[sub]);
      }
    }

    // faithful mask: masked columns get the value 1e-9 (inside softmax)
#pragma unroll
    for (int sub = 0; sub < 4; ++sub) {
      if (smask[sub * 16 + l16] == 0) {
#pragma unroll
        for (int r = 0; r < 8; ++r) sc[sub][r] = 1e-9f;
      }
    }

    // ---- online softmax per row (row = 16 lanes of one half-wave) ----
#pragma unroll
    for (int r = 0; r < 8; ++r) {
      float x0 = sc[0][r], x1 = sc[1][r], x2 = sc[2][r], x3 = sc[3][r];
      float mx = fmaxf(fmaxf(x0, x1), fmaxf(x2, x3));
#pragma unroll
      for (int sh = 8; sh >= 1; sh >>= 1) mx = fmaxf(mx, __shfl_xor(mx, sh, 32));
      float mnew  = fmaxf(mrow[r], mx);
      float alpha = __expf(mrow[r] - mnew);
      float p0 = __expf(x0 - mnew), p1 = __expf(x1 - mnew);
      float p2 = __expf(x2 - mnew), p3 = __expf(x3 - mnew);
      float ps = (p0 + p1) + (p2 + p3);
#pragma unroll
      for (int sh = 8; sh >= 1; sh >>= 1) ps += __shfl_xor(ps, sh, 32);
      lrow[r] = lrow[r] * alpha + ps;
      mrow[r] = mnew;
#pragma unroll
      for (int i = 0; i < 4; ++i) o[i][r] *= alpha;
      // stage P (C-layout -> LDS): row M = r + half*8, col sub*16 + l16
      sP[wave][r + half * 8][l16]      = (__bf16)p0;
      sP[wave][r + half * 8][16 + l16] = (__bf16)p1;
      sP[wave][r + half * 8][32 + l16] = (__bf16)p2;
      sP[wave][r + half * 8][48 + l16] = (__bf16)p3;
    }
    __builtin_amdgcn_wave_barrier();   // wave-local LDS RAW ordering

    // ---- P A-fragments (16x64 -> two 16x32 frags) ----
    v16bf ap[2];
    {
      const unsigned* prow = (const unsigned*)&sP[wave][l16][0];
#pragma unroll
      for (int f = 0; f < 2; ++f)
#pragma unroll
        for (int v = 0; v < 8; ++v)
          ((unsigned*)&ap[f])[v] = prow[(f * 32 + a_kbase(v, half)) >> 1];
    }

    // ---- O += P @ V : 4 dim sub-tiles x 2 key-halves ----
#pragma unroll
    for (int sub = 0; sub < 4; ++sub) {
      const unsigned* rowp = (const unsigned*)&sVt[sub * 16 + l16][0];
#pragma unroll
      for (int f = 0; f < 2; ++f) {
        v16bf bf;
#pragma unroll
        for (int v = 0; v < 8; ++v)
          ((unsigned*)&bf)[v] = rowp[f * 16 + (half << 3) + v];
        o[sub] = wmma_bf16(ap[f], bf, o[sub]);
      }
    }
    __syncthreads();
  }

  // normalize, write merged-head [B,S,E] bf16
#pragma unroll
  for (int sub = 0; sub < 4; ++sub) {
    int d = sub * 16 + l16;
#pragma unroll
    for (int r = 0; r < 8; ++r) {
      int s = q0 + wave * 16 + r + half * 8;
      float val = o[sub][r] / lrow[r];
      O[((size_t)(b * Ssz + s)) * Esz + h * DKsz + d] = (__bf16)val;
    }
  }
}

// ---------------------------------------------------------------------------
// Output projection: out[m,n] = f32( X[m,:] . Wo[n,:] + bo[n] ), X is bf16.
// ---------------------------------------------------------------------------
__global__ __launch_bounds__(256) void out_proj_kernel(
    const __bf16* __restrict__ X, const float* __restrict__ W,
    const float* __restrict__ bias, float* __restrict__ Out) {
  __shared__ __bf16 sX[PBM][PBK];
  __shared__ __bf16 sW[PBN][PBK];

  const int tid  = threadIdx.x;
  const int wave = tid >> 5;
  const int lane = tid & 31;
  const int half = lane >> 4;
  const int l16  = lane & 15;
  const int m0   = blockIdx.x * PBM;
  const int n0   = blockIdx.y * PBN;

  const uint4*  X8 = (const uint4*)X;      // 8 bf16 per uint4, Esz/8 per row
  const float4* W4 = (const float4*)W;

  v8f acc[4] = {};

  for (int k0 = 0; k0 < Esz; k0 += PBK) {
    uint4 xb[4];
#pragma unroll
    for (int i = 0; i < 4; ++i) {           // 128 rows x 8 chunks / 256 thr
      int idx = tid + i * 256;
      int r = idx >> 3, c8 = idx & 7;
      xb[i] = X8[(size_t)(m0 + r) * (Esz / 8) + (k0 >> 3) + c8];
    }
    float4 wb[4];
#pragma unroll
    for (int i = 0; i < 4; ++i) {
      int idx = tid + i * 256;
      int r = idx >> 4, c4 = idx & 15;
      wb[i] = W4[(size_t)(n0 + r) * (Esz / 4) + (k0 >> 2) + c4];
    }
#pragma unroll
    for (int i = 0; i < 4; ++i) {
      int idx = tid + i * 256;
      int r = idx >> 3, c8 = idx & 7;
      *(uint4*)&sX[r][c8 * 8] = xb[i];      // ds_store_b128
    }
#pragma unroll
    for (int i = 0; i < 4; ++i) {
      int idx = tid + i * 256;
      int r = idx >> 4, c4 = idx & 15;
      uint2 pk;
      pk.x = pack_bf16x2(wb[i].x, wb[i].y);
      pk.y = pack_bf16x2(wb[i].z, wb[i].w);
      *(uint2*)&sW[r][c4 * 4] = pk;
    }
    __syncthreads();

    const unsigned* arow = (const unsigned*)&sX[wave * 16 + l16][0];
#pragma unroll
    for (int f = 0; f < 2; ++f) {
      v16bf a;
#pragma unroll
      for (int v = 0; v < 8; ++v)
        ((unsigned*)&a)[v] = arow[(f * 32 + a_kbase(v, half)) >> 1];
#pragma unroll
      for (int sub = 0; sub < 4; ++sub) {
        v16bf bf;
        const unsigned* rowp = (const unsigned*)&sW[sub * 16 + l16][0];
#pragma unroll
        for (int v = 0; v < 8; ++v)
          ((unsigned*)&bf)[v] = rowp[f * 16 + (half << 3) + v];
        acc[sub] = wmma_bf16(a, bf, acc[sub]);
      }
    }
    __syncthreads();
  }

#pragma unroll
  for (int sub = 0; sub < 4; ++sub) {
    int n = n0 + sub * 16 + l16;
    float bv = bias[n];
#pragma unroll
    for (int r = 0; r < 8; ++r) {
      int m = m0 + wave * 16 + r + half * 8;
      Out[(size_t)m * Esz + n] = acc[sub][r] + bv;
    }
  }
}

// ---------------------------------------------------------------------------
extern "C" void kernel_launch(void* const* d_in, const int* in_sizes, int n_in,
                              void* d_out, int out_size, void* d_ws,
                              size_t ws_size, hipStream_t stream) {
  const float* q    = (const float*)d_in[0];
  const float* k    = (const float*)d_in[1];
  const float* v    = (const float*)d_in[2];
  const int*   mask = (const int*)d_in[3];
  const float* wq_w = (const float*)d_in[4];
  const float* wq_b = (const float*)d_in[5];
  const float* wk_w = (const float*)d_in[6];
  const float* wk_b = (const float*)d_in[7];
  const float* wv_w = (const float*)d_in[8];
  const float* wv_b = (const float*)d_in[9];
  const float* wo_w = (const float*)d_in[10];
  const float* wo_b = (const float*)d_in[11];
  float* out = (float*)d_out;

  const size_t NTOK = (size_t)Bsz * Ssz * Esz;   // 4,194,304 elements
  __bf16* Qh = (__bf16*)d_ws;
  __bf16* Kh = Qh + NTOK;
  __bf16* Vh = Kh + NTOK;
  __bf16* Ob = Vh + NTOK;                        // 32 MB bf16 total

  dim3 pg(Bsz * Ssz / PBM, Esz / PBN);           // 32 x 16
  proj_qkv_kernel<<<pg, 256, 0, stream>>>(q, wq_w, wq_b, Qh, 0.125f); // 1/sqrt(64)
  proj_qkv_kernel<<<pg, 256, 0, stream>>>(k, wk_w, wk_b, Kh, 1.0f);
  proj_qkv_kernel<<<pg, 256, 0, stream>>>(v, wv_w, wv_b, Vh, 1.0f);

  dim3 ag(Ssz / 128, Bsz * Hsz);                 // 16 x 32
  attn_kernel<<<ag, 256, 0, stream>>>(Qh, Kh, Vh, mask, Ob);

  out_proj_kernel<<<pg, 256, 0, stream>>>(Ob, wo_w, wo_b, out);
}
